// EMAVectorQuantizer_55353538511028
// MI455X (gfx1250) — matrix operand order repaired and checked by reading, not compiled, not run on hardware
//
#include <hip/hip_runtime.h>
#include <hip/hip_bf16.h>

// ---------------------------------------------------------------------------
// EMA Vector Quantizer for MI455X (gfx1250, wave32, WMMA bf16 + TDM)
// N=32768 rows, D=512 dim, K=4096 codes.
// ---------------------------------------------------------------------------

typedef __attribute__((ext_vector_type(16))) __bf16 v16bf;
typedef __attribute__((ext_vector_type(8)))  __bf16 v8bf;
typedef __attribute__((ext_vector_type(8)))  float  v8f;
typedef __attribute__((ext_vector_type(4)))  unsigned int u32x4;
typedef __attribute__((ext_vector_type(8)))  unsigned int u32x8;

static constexpr int   NN      = 32768;
static constexpr int   DD      = 512;
static constexpr int   KK      = 4096;
static constexpr float DECAY_C = 0.99f;
static constexpr float EPS_C   = 1e-05f;
static constexpr float COMMIT  = 0.25f;

static constexpr int XSTR = DD + 8;   // LDS row stride (halves): bank-conflict free
static constexpr int ESTR = DD + 8;   // TDM reproduces this via pad_amount/interval
static constexpr int TILE_HALVES = 16 * DD;  // 8192 bf16 per embedding slab

// ---------------------------------------------------------------------------
// TDM: DMA one pre-packed 16-column bf16 slab (16*512 halves, contiguous in
// global) into LDS, inserting 16B of padding after every 1024B row so the LDS
// layout has stride ESTR=520 halves.  D# per CDNA5 ISA ch.8.
// ---------------------------------------------------------------------------
__device__ __forceinline__ void tdm_load_tile(const __bf16* gsrc, unsigned ldsOff) {
  unsigned long long ga = (unsigned long long)(uintptr_t)gsrc;
  u32x4 g0;
  g0[0] = 1u;                                      // count=1, user descriptor
  g0[1] = ldsOff;                                  // lds_addr (bytes)
  g0[2] = (unsigned)ga;                            // global_addr[31:0]
  g0[3] = ((unsigned)(ga >> 32) & 0x01FFFFFFu)     // global_addr[56:32]
          | 0x80000000u;                           // type=2 ("image")
  u32x8 g1;
  // data_size=1 (2B), pad_enable, pad_interval=7 (256 DW), pad_amount=3 (4 DW)
  g1[0] = (1u << 16) | (1u << 20) | (7u << 22) | (3u << 25);
  g1[1] = 0x20000000u;                             // tensor_dim0[15:0]=8192 @bits63:48
  g1[2] = 0x00010000u;                             // tensor_dim1=1 @bits95:80
  g1[3] = 0x20000000u;                             // tile_dim0=8192 @bits127:112
  g1[4] = 0x00000001u;                             // tile_dim1=1, tile_dim2=0
  g1[5] = 0x00002000u;                             // tensor_dim0_stride=8192
  g1[6] = 0x20000000u;                             // tensor_dim1_stride[15:0]=8192
  g1[7] = 0u;
  asm volatile("tensor_load_to_lds %0, %1" :: "s"(g0), "s"(g1) : "memory");
}

// ---------------------------------------------------------------------------
// Workspace zeroing (counts, loss/n accumulators, embed_sum)
// ---------------------------------------------------------------------------
__global__ void vq_zero_kernel(float* __restrict__ p, size_t cnt) {
  size_t i = (size_t)blockIdx.x * blockDim.x + threadIdx.x;
  size_t stride = (size_t)gridDim.x * blockDim.x;
  for (; i < cnt; i += stride) p[i] = 0.0f;
}

// ---------------------------------------------------------------------------
// Prep: embT[k][d] = emb[d][k] (coalesced gather later), eNorm[k] = sum_d e^2
// ---------------------------------------------------------------------------
__global__ __launch_bounds__(256)
void vq_prep_kernel(const float* __restrict__ emb,
                    float* __restrict__ embT, float* __restrict__ eNorm) {
  int k = blockIdx.x * 256 + threadIdx.x;
  float s = 0.0f;
  for (int d = 0; d < DD; ++d) {
    float v = emb[(size_t)d * KK + k];   // coalesced across k
    embT[(size_t)k * DD + d] = v;
    s += v * v;
  }
  eNorm[k] = s;
}

// ---------------------------------------------------------------------------
// Pack embedding into bf16 TDM slabs: ePack[kt][n][d] = emb[d][k0+n].
// One block per 16-column tile; writes fully coalesced, reads L2-resident.
// ---------------------------------------------------------------------------
__global__ __launch_bounds__(256)
void vq_pack_kernel(const float* __restrict__ emb, __bf16* __restrict__ ePack) {
  const int kt = blockIdx.x;
  const int k0 = kt * 16;
  for (int j = threadIdx.x; j < TILE_HALVES; j += 256) {
    int n = j >> 9;          // 0..15
    int d = j & (DD - 1);
    ePack[(size_t)kt * TILE_HALVES + j] = (__bf16)emb[(size_t)d * KK + k0 + n];
  }
}

// ---------------------------------------------------------------------------
// Fused distance GEMM + row argmin.
// 128 threads (4 waves) / 64 rows per workgroup; x staged once as bf16 in LDS.
// B slabs arrive via double-buffered TENSOR_LOAD_TO_LDS overlapped with the
// 16 V_WMMA_F32_16X16X32_BF16 per tile; argmin fused via half-group butterfly.
// ---------------------------------------------------------------------------
__global__ __launch_bounds__(128)
void vq_argmin_kernel(const float* __restrict__ x, const __bf16* __restrict__ ePack,
                      const float* __restrict__ eNorm, int* __restrict__ idxOut) {
  __shared__ alignas(16) __bf16 xs[64 * XSTR];        // 64 rows x 512 (bf16)
  __shared__ alignas(16) __bf16 esT[2][16 * ESTR];    // double-buffered slab

  const int tid  = threadIdx.x;
  const int lane = tid & 31;
  const int wave = tid >> 5;
  const int row0 = blockIdx.x * 64;
  constexpr int NT = KK / 16;

  // Stage x block as bf16 (coalesced: d fastest)
  for (int i = tid; i < 64 * DD; i += 128) {
    int r = i >> 9;
    int d = i & (DD - 1);
    xs[r * XSTR + d] = (__bf16)x[(size_t)(row0 + r) * DD + d];
  }

  // Kick off first slab DMA from wave 0 while others finish staging x.
  if (wave == 0) {
    tdm_load_tile(&ePack[0], (unsigned)(uintptr_t)&esT[0][0]);
    __builtin_amdgcn_s_wait_tensorcnt(0);
  }
  __syncthreads();

  const int n  = lane & 15;      // column within tile (D-matrix N index)
  const int hi = lane >> 4;      // half-wave select
  const int ar = wave * 16 + n;  // A-matrix row for this lane

  float best[8];
  int   bidx[8];
#pragma unroll
  for (int v = 0; v < 8; ++v) { best[v] = 3.4e38f; bidx[v] = 0; }

  for (int kt = 0; kt < NT; ++kt) {
    const int k0 = kt * 16;

    // DMA next slab into the other buffer, overlapped with this tile's WMMAs.
    if (wave == 0 && kt + 1 < NT)
      tdm_load_tile(&ePack[(size_t)(kt + 1) * TILE_HALVES],
                    (unsigned)(uintptr_t)&esT[(kt + 1) & 1][0]);

    if (tid < 64 && kt + 2 < NT)   // warm L2 two tiles ahead
      __builtin_prefetch(&ePack[(size_t)(kt + 2) * TILE_HALVES + tid * 128], 0, 1);

    v8f c = {};  // f32 accumulator (16x16 C/D fragment)
    const __bf16* ap_base = &xs[ar * XSTR];
    const __bf16* bp_base = &esT[kt & 1][n * ESTR];
#pragma unroll
    for (int dsi = 0; dsi < DD / 32; ++dsi) {
      const int d0 = dsi * 32;
      v16bf a, b;
      // A 16x32 bf16 layout: lanes<16 hold K 0-7 & 16-23; lanes>=16 hold 8-15 & 24-31
      const __bf16* ap = ap_base + d0 + (hi ? 8 : 0);
      ((v8bf*)&a)[0] = *(const v8bf*)(ap);
      ((v8bf*)&a)[1] = *(const v8bf*)(ap + 16);
      // B 32x16 bf16 layout: lane = n; lanes<16 hold K 0-7 & 16-23; lanes>=16 hold 8-15 & 24-31
      const __bf16* bp = bp_base + d0 + (hi ? 8 : 0);
      ((v8bf*)&b)[0] = *(const v8bf*)(bp);
      ((v8bf*)&b)[1] = *(const v8bf*)(bp + 16);
      c = __builtin_amdgcn_wmma_f32_16x16x32_bf16(false, a, false, b,
                                                  (short)0, c, false, false);
    }

    // score = ||e||^2 - 2*dot ; argmin butterfly across the 16-lane half-group
    const float en = eNorm[k0 + n];
#pragma unroll
    for (int v = 0; v < 8; ++v) {
      float s = en - 2.0f * c[v];
      int   bi = k0 + n;
#pragma unroll
      for (int off = 1; off <= 8; off <<= 1) {
        float os = __shfl_xor(s, off, 32);
        int   oi = __shfl_xor(bi, off, 32);
        if (os < s || (os == s && oi < bi)) { s = os; bi = oi; }
      }
      if (s < best[v] || (s == best[v] && bi < bidx[v])) { best[v] = s; bidx[v] = bi; }
    }

    // Wait for the overlapped DMA, then publish the buffer to all waves.
    if (wave == 0 && kt + 1 < NT) __builtin_amdgcn_s_wait_tensorcnt(0);
    __syncthreads();
  }

  // C/D layout: VGPR v holds row v (lanes 0-15) and row v+8 (lanes 16-31)
  if (n == 0) {
    int rbase = row0 + wave * 16 + hi * 8;
#pragma unroll
    for (int v = 0; v < 8; ++v) idxOut[rbase + v] = bidx[v];
  }
}

// ---------------------------------------------------------------------------
// Gather quantized rows, scatter embed_sum (segment_sum), counts, loss accum.
// ---------------------------------------------------------------------------
__global__ __launch_bounds__(128)
void vq_gather_kernel(const float* __restrict__ x, const float* __restrict__ embT,
                      const int* __restrict__ idxIn,
                      float* __restrict__ qOut, float* __restrict__ idxF,
                      float* __restrict__ embed_sum, float* __restrict__ counts,
                      float* __restrict__ lossAcc) {
  __shared__ float red[128];
  const int row = blockIdx.x;
  const int idx = idxIn[row];
  float acc = 0.0f;
#pragma unroll
  for (int j = 0; j < DD / 128; ++j) {
    int d = threadIdx.x + j * 128;
    float q  = embT[(size_t)idx * DD + d];
    float xv = x[(size_t)row * DD + d];
    qOut[(size_t)row * DD + d] = q;           // straight-through forward == quantized
    atomicAdd(&embed_sum[(size_t)idx * DD + d], xv);
    float df = q - xv;
    acc += df * df;
  }
  red[threadIdx.x] = acc;
  __syncthreads();
  for (int s = 64; s > 0; s >>= 1) {
    if (threadIdx.x < s) red[threadIdx.x] += red[threadIdx.x + s];
    __syncthreads();
  }
  if (threadIdx.x == 0) {
    atomicAdd(lossAcc, red[0]);
    atomicAdd(&counts[idx], 1.0f);
    idxF[row] = (float)idx;
  }
}

// ---------------------------------------------------------------------------
// new_cluster_size = cs*decay + (1-decay)*counts ; n = sum(new_cluster_size)
// ---------------------------------------------------------------------------
__global__ __launch_bounds__(256)
void vq_cluster_kernel(const float* __restrict__ cs, const float* __restrict__ counts,
                       float* __restrict__ ncsOut, float* __restrict__ nAcc) {
  __shared__ float red[256];
  int k = blockIdx.x * 256 + threadIdx.x;
  float ncs = cs[k] * DECAY_C + (1.0f - DECAY_C) * counts[k];
  ncsOut[k] = ncs;
  red[threadIdx.x] = ncs;
  __syncthreads();
  for (int s = 128; s > 0; s >>= 1) {
    if (threadIdx.x < s) red[threadIdx.x] += red[threadIdx.x + s];
    __syncthreads();
  }
  if (threadIdx.x == 0) atomicAdd(nAcc, red[0]);
}

// ---------------------------------------------------------------------------
// new_ema_w = ema_w*decay + (1-decay)*embed_sum.T ; new_embedding = ema/smoothed
// ---------------------------------------------------------------------------
__global__ __launch_bounds__(256)
void vq_finalize_kernel(const float* __restrict__ emaw, const float* __restrict__ embed_sum,
                        const float* __restrict__ ncs, const float* __restrict__ nAcc,
                        const float* __restrict__ lossAcc,
                        float* __restrict__ emaOut, float* __restrict__ embOut,
                        float* __restrict__ lossOut) {
  __shared__ float t[16][17];
  const int tx = threadIdx.x, ty = threadIdx.y;
  const int k0 = blockIdx.x * 16, d0 = blockIdx.y * 16;
  t[ty][tx] = embed_sum[(size_t)(k0 + ty) * DD + d0 + tx];  // coalesced in d
  __syncthreads();
  const int k = k0 + tx, d = d0 + ty;
  float es = t[tx][ty];  // embed_sum[k][d]
  float ne = emaw[(size_t)d * KK + k] * DECAY_C + (1.0f - DECAY_C) * es;
  emaOut[(size_t)d * KK + k] = ne;
  float nv = *nAcc;
  float sm = (ncs[k] + EPS_C) / (nv + (float)KK * EPS_C) * nv;
  embOut[(size_t)d * KK + k] = ne / sm;
  if (blockIdx.x == 0 && blockIdx.y == 0 && tx == 0 && ty == 0)
    *lossOut = COMMIT * (*lossAcc) / ((float)NN * (float)DD);
}

// ---------------------------------------------------------------------------
extern "C" void kernel_launch(void* const* d_in, const int* in_sizes, int n_in,
                              void* d_out, int out_size, void* d_ws, size_t ws_size,
                              hipStream_t stream) {
  const float* x    = (const float*)d_in[0];  // [N, D]
  const float* emb  = (const float*)d_in[1];  // [D, K]
  const float* cs   = (const float*)d_in[2];  // [K]
  const float* emaw = (const float*)d_in[3];  // [D, K]

  // Output layout (tuple, flattened in return order)
  float* out     = (float*)d_out;
  float* qOut    = out;                              // N*D
  float* lossOut = qOut + (size_t)NN * DD;           // 1
  float* idxF    = lossOut + 1;                      // N
  float* embOut  = idxF + NN;                        // D*K
  float* ncsOut  = embOut + (size_t)DD * KK;         // K
  float* emaOut  = ncsOut + KK;                      // D*K

  // Workspace: [counts K][lossAcc][nAcc][embed_sum K*D] <- zeroed region
  //            [eNorm K][idx N(int)][embT K*D (f32)][ePack K*D (bf16)]
  float* ws        = (float*)d_ws;
  float* counts    = ws;
  float* lossAcc   = counts + KK;
  float* nAcc      = lossAcc + 1;
  float* embed_sum = nAcc + 1;
  float* eNorm     = embed_sum + (size_t)KK * DD;
  int*   idxW      = (int*)(eNorm + KK);
  float* embT      = (float*)(idxW + NN);
  __bf16* ePack    = (__bf16*)(embT + (size_t)KK * DD);

  size_t zcnt = (size_t)KK + 2 + (size_t)KK * DD;
  vq_zero_kernel<<<2048, 256, 0, stream>>>(ws, zcnt);
  vq_prep_kernel<<<KK / 256, 256, 0, stream>>>(emb, embT, eNorm);
  vq_pack_kernel<<<KK / 16, 256, 0, stream>>>(emb, ePack);
  vq_argmin_kernel<<<NN / 64, 128, 0, stream>>>(x, ePack, eNorm, idxW);
  vq_gather_kernel<<<NN, 128, 0, stream>>>(x, embT, idxW, qOut, idxF,
                                           embed_sum, counts, lossAcc);
  vq_cluster_kernel<<<KK / 256, 256, 0, stream>>>(cs, counts, ncsOut, nAcc);
  dim3 g2(KK / 16, DD / 16), b2(16, 16);
  vq_finalize_kernel<<<g2, b2, 0, stream>>>(emaw, embed_sum, ncsOut, nAcc, lossAcc,
                                            emaOut, embOut, lossOut);
}
